// Alpha_Free_GLM_18021682774404
// MI455X (gfx1250) — matrix operand order repaired and checked by reading, not compiled, not run on hardware
//
#include <hip/hip_runtime.h>
#include <hip/hip_bf16.h>
#include <math.h>

typedef __attribute__((ext_vector_type(16))) _Float16 v16h;
typedef __attribute__((ext_vector_type(8)))  float    v8f;

#define TDATA   100000
#define SUBN    20
#define TNO     201
#define ENO     500
#define INO     100
#define FILT_ELEMS (2 * SUBN * TNO)   // 8040
#define TT      200                   // time tile for conv kernel

__device__ __forceinline__ v8f wmma_f16(v16h a, v16h b, v8f c) {
  // D = A(16x32 f16) * B(32x16 f16) + C(16x16 f32)
  return __builtin_amdgcn_wmma_f32_16x16x32_f16(
      /*neg_a=*/false, a, /*neg_b=*/false, b,
      /*c_mod=*/(short)0, c, /*reuse_a=*/false, /*reuse_b=*/false);
}

// ---------------------------------------------------------------------------
// Pre-pack B = Csyn^T (contraction over k of Csyn[n][k]) into the exact
// 16-bit B 32x16 WMMA fragment layout, zero-padded for n>=20 and k>=K:
//   Bpack[((nt*KB + kb)*32 + lane)*16 + h] = Csyn[n][k]
//   n = nt*16 + lane%16,  k = kb*32 + (h<8 ? (lane>=16 ? 8 : 0) + h : (lane>=16 ? 24 : 16) + h-8)
// Each lane's 16 halves are one contiguous 32-byte v16h fragment.
// ---------------------------------------------------------------------------
__global__ void pack_B(const float* __restrict__ Csyn, _Float16* __restrict__ Bpack,
                       int K, int KB, int total)
{
  const int idx = blockIdx.x * 256 + threadIdx.x;
  if (idx >= total) return;
  const int h    = idx & 15;
  const int lane = (idx >> 4) & 31;
  const int q    = idx >> 9;          // nt*KB + kb
  const int kb   = q % KB;
  const int nt   = q / KB;
  const int n    = nt * 16 + (lane & 15);
  const int k    = kb * 32 + (h < 8 ? (lane >= 16 ? 8 : 0) + h : (lane >= 16 ? 24 : 16) + (h - 8));
  const float v  = (n < SUBN && k < K) ? Csyn[(size_t)n * K + k] : 0.0f;
  Bpack[idx] = (_Float16)v;
}

// ---------------------------------------------------------------------------
// proj[t*40 + outOff + n] = sum_k S[t*K + k] * Csyn[n*K + k]   (n = 0..19)
// S entries are 0/1 -> exact in f16; f32 accumulation via WMMA.
// One 16-row M tile per wave; B fragments come pre-packed (no divergence).
// ---------------------------------------------------------------------------
template<int K>
__global__ __launch_bounds__(128)
void proj_gemm(const float* __restrict__ S, const _Float16* __restrict__ Bpack,
               float* __restrict__ proj, int T, int outOff)
{
  constexpr int KB    = (K + 31) / 32;   // total K tiles (incl. tail)
  constexpr int KFULL = K / 32;          // full 32-wide K tiles

  const int wave = threadIdx.x >> 5;
  const int lane = threadIdx.x & 31;
  const int l15  = lane & 15;
  const bool hi  = lane >= 16;

  const int mtile = blockIdx.x * 4 + wave;
  if (mtile * 16 >= T) return;           // wave-uniform, EXEC stays full
  const int m0 = mtile * 16;

  const float* Arow = S + (size_t)(m0 + l15) * K;
  const v16h*  Bp   = (const v16h*)Bpack;           // [nt][KB][32 lanes]
  const v16h*  Bp0  = Bp + (size_t)lane;            // nt = 0
  const v16h*  Bp1  = Bp + (size_t)KB * 32 + lane;  // nt = 1

  v8f acc0 = {};
  v8f acc1 = {};

#pragma unroll 4
  for (int kb = 0; kb < KFULL; ++kb) {
    const int kb32 = kb * 32;
    // A fragment (16-bit A 16x32 layout): lanes 0-15 K {0..7,16..23},
    // lanes 16-31 K {8..15,24..31}, row = m0 + lane%16.
    const int ka1 = kb32 + (hi ? 8 : 0);
    const int ka2 = kb32 + (hi ? 24 : 16);
    __builtin_prefetch(Arow + kb32 + 128, 0, 1);    // global_prefetch_b8
    const float4 a0 = *(const float4*)(Arow + ka1);
    const float4 a1 = *(const float4*)(Arow + ka1 + 4);
    const float4 a2 = *(const float4*)(Arow + ka2);
    const float4 a3 = *(const float4*)(Arow + ka2 + 4);
    v16h A;
    A[0]=(_Float16)a0.x;  A[1]=(_Float16)a0.y;  A[2]=(_Float16)a0.z;  A[3]=(_Float16)a0.w;
    A[4]=(_Float16)a1.x;  A[5]=(_Float16)a1.y;  A[6]=(_Float16)a1.z;  A[7]=(_Float16)a1.w;
    A[8]=(_Float16)a2.x;  A[9]=(_Float16)a2.y;  A[10]=(_Float16)a2.z; A[11]=(_Float16)a2.w;
    A[12]=(_Float16)a3.x; A[13]=(_Float16)a3.y; A[14]=(_Float16)a3.z; A[15]=(_Float16)a3.w;

    const v16h B0 = Bp0[kb * 32];
    const v16h B1 = Bp1[kb * 32];
    acc0 = wmma_f16(A, B0, acc0);
    acc1 = wmma_f16(A, B1, acc1);
  }

  if constexpr ((K % 32) != 0) {         // tail K tile: branchless clamped loads
    const int kb32 = KFULL * 32;
    v16h A;
#pragma unroll
    for (int h = 0; h < 16; ++h) {
      const int k = kb32 + (h < 8 ? (hi ? 8 : 0) + h : (hi ? 24 : 16) + (h - 8));
      const float x = Arow[k < K ? k : (K - 1)];    // always in-bounds
      A[h] = (k < K) ? (_Float16)x : (_Float16)0.0f; // v_cndmask, no exec games
    }
    const v16h B0 = Bp0[KFULL * 32];
    const v16h B1 = Bp1[KFULL * 32];
    acc0 = wmma_f16(A, B0, acc0);
    acc1 = wmma_f16(A, B1, acc1);
  }

  // D layout: VGPR r = row m0+r (lanes 0-15) / m0+8+r (lanes 16-31), col = lane%16
#pragma unroll
  for (int r = 0; r < 8; ++r) {
    const int mr = m0 + r + (hi ? 8 : 0);
    float* dst = proj + (size_t)mr * 40 + outOff;
    dst[l15] = acc0[r];
    if (l15 < SUBN - 16) dst[16 + l15] = acc1[r];   // cols 16..19 of second tile
  }
}

// syn_kern = syn_weights(40x3) @ syn_basis(3x201).
// out_filters == syn_kern exactly (double flip cancels); also store flipped copy.
__global__ void make_kern(const float* __restrict__ sw, const float* __restrict__ sb,
                          float* __restrict__ kflip, float* __restrict__ outF)
{
  const int idx = blockIdx.x * 256 + threadIdx.x;
  if (idx >= FILT_ELEMS) return;
  const int r = idx / TNO;
  const int t = idx - r * TNO;
  const float v = sw[r*3 + 0] * sb[t]
                + sw[r*3 + 1] * sb[TNO + t]
                + sw[r*3 + 2] * sb[2*TNO + t];
  outF[idx] = v;                         // out_filters region of d_out
  kflip[r * TNO + (TNO - 1 - t)] = v;    // time-reversed for causal FIR
}

// Per time-tile: 201-tap depthwise FIR over 40 channels (LDS halo), then the
// 20-node dendritic tanh tree (fully unrolled, binary-tree children 2s+1,2s+2).
__global__ __launch_bounds__(256)
void conv_tree(const float* __restrict__ proj, const float* __restrict__ kflip,
               const float* __restrict__ Wsub, const float* __restrict__ Theta,
               const float* __restrict__ Vo, float* __restrict__ out, int T)
{
  __shared__ float P[40][TT + TNO + 7];          // 40 x 408 floats = 65280 B
  const int t0  = blockIdx.x * TT;
  const int tid = threadIdx.x;

  // cooperative halo load: rows t0-200 .. t0+TT-1, channel-major in LDS
  for (int idx = tid; idx < (TT + TNO - 1) * 40; idx += 256) {
    const int row = idx / 40;
    const int ch  = idx - row * 40;
    const int t   = t0 - (TNO - 1) + row;
    P[ch][row] = (t >= 0 && t < T) ? proj[(size_t)t * 40 + ch] : 0.0f;
  }
  __syncthreads();

  const int t = t0 + tid;
  if (tid >= TT || t >= T) return;

  float syn[SUBN];
#pragma unroll
  for (int s = 0; s < SUBN; ++s) {
    const float* pe = &P[s][tid];                // lanes hit consecutive banks
    const float* pi = &P[s + SUBN][tid];
    const float* ke = kflip + s * TNO;           // wave-uniform -> scalar loads
    const float* ki = kflip + (s + SUBN) * TNO;
    float acc = 0.0f;
    for (int j = 0; j < TNO; ++j)
      acc += pe[j] * ke[j] + pi[j] * ki[j];
    syn[s] = acc;
  }

  float ew[SUBN];
#pragma unroll
  for (int s = 0; s < SUBN; ++s) ew[s] = expf(Wsub[s]);

  float col[SUBN];
#pragma unroll
  for (int s = SUBN - 1; s >= 0; --s) {
    float a = syn[s] + Theta[s];
    if (2*s + 1 < SUBN) a += col[2*s + 1] * ew[2*s + 1];
    if (2*s + 2 < SUBN) a += col[2*s + 2] * ew[2*s + 2];
    col[s] = tanhf(a);
  }
  out[t] = col[0] * ew[0] + Vo[0];
}

extern "C" void kernel_launch(void* const* d_in, const int* in_sizes, int n_in,
                              void* d_out, int out_size, void* d_ws, size_t ws_size,
                              hipStream_t stream) {
  const float* S_e   = (const float*)d_in[0];   // (100000, 500)
  const float* S_i   = (const float*)d_in[1];   // (100000, 100)
  const float* C_e   = (const float*)d_in[2];   // (20, 500)
  const float* C_i   = (const float*)d_in[3];   // (20, 100)
  const float* sw    = (const float*)d_in[4];   // (40, 3)
  const float* sb    = (const float*)d_in[5];   // (3, 201)
  const float* Wsub  = (const float*)d_in[6];   // (20,)
  const float* Theta = (const float*)d_in[7];   // (20,)
  const float* Vo    = (const float*)d_in[8];   // (1,)

  float* out  = (float*)d_out;                  // [V(100000) | filt(8040) | Ce(10000) | Ci(2000)]
  float* proj = (float*)d_ws;                   // TDATA x 40 f32 (e: 0-19, i: 20-39)
  float* kflip = proj + (size_t)TDATA * 40;     // 8040 f32, time-reversed kernels
  constexpr int KB_E = (ENO + 31) / 32;         // 16
  constexpr int KB_I = (INO + 31) / 32;         // 4
  constexpr int BPE_ELEMS = 2 * KB_E * 32 * 16; // 16384 halves (32 KB)
  constexpr int BPI_ELEMS = 2 * KB_I * 32 * 16; // 4096 halves (8 KB)
  _Float16* BpackE = (_Float16*)(kflip + FILT_ELEMS);        // 32B-aligned
  _Float16* BpackI = BpackE + BPE_ELEMS;                     // 32B-aligned

  pack_B<<<(BPE_ELEMS + 255) / 256, 256, 0, stream>>>(C_e, BpackE, ENO, KB_E, BPE_ELEMS);
  pack_B<<<(BPI_ELEMS + 255) / 256, 256, 0, stream>>>(C_i, BpackI, INO, KB_I, BPI_ELEMS);
  make_kern<<<(FILT_ELEMS + 255) / 256, 256, 0, stream>>>(sw, sb, kflip, out + TDATA);

  const int mtiles  = TDATA / 16;               // 6250
  const int gblocks = (mtiles + 3) / 4;         // 4 waves (M-tiles) per 128-thread block
  proj_gemm<ENO><<<gblocks, 128, 0, stream>>>(S_e, BpackE, proj, TDATA, 0);
  proj_gemm<INO><<<gblocks, 128, 0, stream>>>(S_i, BpackI, proj, TDATA, SUBN);

  conv_tree<<<(TDATA + TT - 1) / TT, 256, 0, stream>>>(proj, kflip, Wsub, Theta, Vo, out, TDATA);

  hipMemcpyAsync(out + TDATA + FILT_ELEMS,           C_e, (size_t)SUBN * ENO * sizeof(float),
                 hipMemcpyDeviceToDevice, stream);
  hipMemcpyAsync(out + TDATA + FILT_ELEMS + SUBN*ENO, C_i, (size_t)SUBN * INO * sizeof(float),
                 hipMemcpyDeviceToDevice, stream);
}
